// BossBD2Model_77137612636758
// MI455X (gfx1250) — matrix operand order repaired and checked by reading, not compiled, not run on hardware
//
#include <hip/hip_runtime.h>
#include <stdint.h>

typedef _Float16 v16h __attribute__((ext_vector_type(16)));
typedef float    v8f  __attribute__((ext_vector_type(8)));
typedef float    v2f  __attribute__((ext_vector_type(2)));
typedef uint32_t v8u  __attribute__((ext_vector_type(8)));

#define LOG2E  1.4426950408889634f
#define LOG2E2 2.8853900817779268f

__device__ __forceinline__ float fast_sig(float x) {
    // 1 / (1 + 2^(-x*log2e))  -> v_exp_f32 + v_rcp_f32
    float e = __builtin_amdgcn_exp2f(-LOG2E * x);
    return __builtin_amdgcn_rcpf(1.0f + e);
}
__device__ __forceinline__ float fast_tanh(float x) {
    // tanh(x) = 2/(1+2^(-2x*log2e)) - 1 ; saturates correctly at +-1
    float e = __builtin_amdgcn_exp2f(-LOG2E2 * x);
    return __builtin_amdgcn_rcpf(1.0f + e) * 2.0f - 1.0f;
}

__device__ __forceinline__ v8f wmma_pre(v2f a, v2f b) {
    v8f z; 
    #pragma unroll
    for (int i = 0; i < 8; ++i) z[i] = 0.0f;
    return __builtin_amdgcn_wmma_f32_16x16x4_f32(false, a, false, b,
                                                 (short)0, z, false, false);
}
__device__ __forceinline__ v8f wmma_hh(v16h a, v16h b, v8f c) {
    return __builtin_amdgcn_wmma_f32_16x16x32_f16(false, a, false, b,
                                                  (short)0, c, false, false);
}
__device__ __forceinline__ uint32_t swapx16_u32(uint32_t v) {
    // group-of-32 swizzle: and=0x1f, or=0, xor=0x10  -> lane l <-> l^16
    return (uint32_t)__builtin_amdgcn_ds_swizzle((int)v, 0x401f);
}

// One wave per 16-batch half; everything lives in registers across all T steps.
__global__ __launch_bounds__(32) void lstm_seq_kernel(
    const float* __restrict__ x,     // (B=32, T) flat
    const float* __restrict__ W_ih,  // (128, 1)
    const float* __restrict__ W_hh,  // (128, 32)
    const float* __restrict__ b_ih,  // (128)
    const float* __restrict__ b_hh,  // (128)
    const float* __restrict__ W_out, // (1, 32)
    const float* __restrict__ b_out, // (1)
    float* __restrict__ out,         // (B=32, T) flat
    int T)
{
    const int l    = threadIdx.x;      // 0..31
    const int lo16 = l & 15;
    const int hi   = l >> 4;           // lane half: selects M/K sub-block
    const int wave = blockIdx.x;       // 0..1 -> batch halves
    const int b    = lo16 + 16 * wave; // batch column owned by this lane

    // ---- Preload W_hh tiles in f16 A-layout (8 tiles of 16 gates x 32 K) ----
    // A 16x32 f16 (k order used here, B matches): lanes 0-15: M=l, K 0-15; lanes 16-31: M=l-16, K 16-31
    v16h Ahh[8];
    const int k0 = hi ? 16 : 0;
    #pragma unroll
    for (int n = 0; n < 8; ++n) {
        const float* row = W_hh + (16 * n + lo16) * 32;
        v16h a;
        #pragma unroll
        for (int i = 0; i < 8; ++i) {
            a[i]     = (_Float16)row[k0 + i];
            a[8 + i] = (_Float16)row[k0 + 8 + i];
        }
        Ahh[n] = a;
    }
    // ---- Preload [W_ih | (b_ih+b_hh)] as f32 16x4 A tiles (K=0: W_ih, K=1: bias) ----
    v2f Aih[8];
    #pragma unroll
    for (int n = 0; n < 8; ++n) {
        int g = 16 * n + lo16;
        v2f a;
        a[0] = hi ? 0.0f : W_ih[g];
        a[1] = hi ? 0.0f : (b_ih[g] + b_hh[g]);
        Aih[n] = a;
    }
    // ---- Per-lane W_out slices (this lane's hidden indices) ----
    float woA[8], woB[8];
    #pragma unroll
    for (int r = 0; r < 8; ++r) {
        woA[r] = W_out[r + 8 * hi];
        woB[r] = W_out[16 + r + 8 * hi];
    }
    const float bo = b_out[0];

    // ---- State: c in f32 (C-layout positions), h as f16 B-operand ----
    float cA[8], cB[8];
    #pragma unroll
    for (int r = 0; r < 8; ++r) { cA[r] = 0.0f; cB[r] = 0.0f; }
    v16h hB;
    #pragma unroll
    for (int i = 0; i < 16; ++i) hB[i] = (_Float16)0.0f;

    const size_t xbase = (size_t)b * (size_t)T;
    float xv = x[xbase];  // t = 0

    for (int t = 0; t < T; ++t) {
        // Prefetch next x early (double buffer); latency hides under the step.
        int tn = (t + 1 < T) ? (t + 1) : t;
        float xnext = x[xbase + tn];

        // B for the rank-2 pre-gate WMMA: K=0 -> x[b], K=1 -> 1.0 (bias), K=2,3 -> 0
        v2f Bx;
        Bx[0] = hi ? 0.0f : xv;
        Bx[1] = hi ? 0.0f : 1.0f;

        // ---- Gates for hidden j in [0,16): tiles i=0, f=2, g=4, o=6 ----
        v8f gI = wmma_hh(Ahh[0], hB, wmma_pre(Aih[0], Bx));
        v8f gF = wmma_hh(Ahh[2], hB, wmma_pre(Aih[2], Bx));
        v8f gG = wmma_hh(Ahh[4], hB, wmma_pre(Aih[4], Bx));
        v8f gO = wmma_hh(Ahh[6], hB, wmma_pre(Aih[6], Bx));
        float hvA[8];
        #pragma unroll
        for (int r = 0; r < 8; ++r) {
            float c = fast_sig(gF[r]) * cA[r] + fast_sig(gI[r]) * fast_tanh(gG[r]);
            cA[r] = c;
            hvA[r] = fast_sig(gO[r]) * fast_tanh(c);
        }
        // ---- Gates for hidden j in [16,32): tiles i=1, f=3, g=5, o=7 ----
        v8f gI2 = wmma_hh(Ahh[1], hB, wmma_pre(Aih[1], Bx));
        v8f gF2 = wmma_hh(Ahh[3], hB, wmma_pre(Aih[3], Bx));
        v8f gG2 = wmma_hh(Ahh[5], hB, wmma_pre(Aih[5], Bx));
        v8f gO2 = wmma_hh(Ahh[7], hB, wmma_pre(Aih[7], Bx));
        float hvB[8];
        #pragma unroll
        for (int r = 0; r < 8; ++r) {
            float c = fast_sig(gF2[r]) * cB[r] + fast_sig(gI2[r]) * fast_tanh(gG2[r]);
            cB[r] = c;
            hvB[r] = fast_sig(gO2[r]) * fast_tanh(c);
        }

        // ---- Rebuild h^T as f16 B-layout via pack + lane-pair swap (SWAPX16) ----
        // lane(hi=0) owns j=r,16+r ; lane(hi=1) owns j=8+r,24+r ; partner supplies the rest.
        uint32_t pkA[4], pkB[4];
        #pragma unroll
        for (int v = 0; v < 4; ++v) {
            pkA[v] = __builtin_bit_cast(uint32_t,
                        __builtin_amdgcn_cvt_pkrtz(hvA[2 * v], hvA[2 * v + 1]));
            pkB[v] = __builtin_bit_cast(uint32_t,
                        __builtin_amdgcn_cvt_pkrtz(hvB[2 * v], hvB[2 * v + 1]));
        }
        v8u bu;
        #pragma unroll
        for (int v = 0; v < 4; ++v) {
            uint32_t send = hi ? pkA[v] : pkB[v];
            uint32_t got  = swapx16_u32(send);
            bu[v]     = hi ? got    : pkA[v];  // K 0-7   / 16-23
            bu[4 + v] = hi ? pkB[v] : got;     // K 8-15  / 24-31
        }
        hB = __builtin_bit_cast(v16h, bu);

        // ---- Output projection: out[b,t] = sum_j h[b,j]*W_out[j] + b_out ----
        float p = 0.0f;
        #pragma unroll
        for (int r = 0; r < 8; ++r) p += hvA[r] * woA[r] + hvB[r] * woB[r];
        float pp = __builtin_bit_cast(float, swapx16_u32(__builtin_bit_cast(uint32_t, p)));
        // lane pairs (l, l^16) hold complementary hidden halves of the same batch:
        // both compute bitwise-identical totals -> duplicate store is benign, EXEC stays full.
        out[xbase + t] = p + pp + bo;

        xv = xnext;
    }
}

extern "C" void kernel_launch(void* const* d_in, const int* in_sizes, int n_in,
                              void* d_out, int out_size, void* d_ws, size_t ws_size,
                              hipStream_t stream) {
    const float* x     = (const float*)d_in[0];
    const float* W_ih  = (const float*)d_in[1];
    const float* W_hh  = (const float*)d_in[2];
    const float* b_ih  = (const float*)d_in[3];
    const float* b_hh  = (const float*)d_in[4];
    const float* W_out = (const float*)d_in[5];
    const float* b_out = (const float*)d_in[6];
    float* out = (float*)d_out;
    const int B = 32;
    const int T = in_sizes[0] / B;  // x has B*T elements

    // Two independent single-wave workgroups, one per 16-batch half.
    lstm_seq_kernel<<<2, 32, 0, stream>>>(x, W_ih, W_hh, b_ih, b_hh,
                                          W_out, b_out, out, T);
}
